// AdaptiveInterpolationModule_63685775065588
// MI455X (gfx1250) — matrix operand order repaired and checked by reading, not compiled, not run on hardware
//
#include <hip/hip_runtime.h>

typedef __attribute__((ext_vector_type(16))) _Float16 v16h;
typedef __attribute__((ext_vector_type(8)))  float    v8f;

#define DIM 768
#define SEQ 1024
#define NBATCH 16
#define NTOK (NBATCH * SEQ)

static constexpr float SM_SCALE = 0.03608439182435161f; // 1/sqrt(768)

__device__ __forceinline__ _Float16 h_lo(unsigned r) {
  return __builtin_bit_cast(_Float16, (unsigned short)(r & 0xffffu));
}
__device__ __forceinline__ _Float16 h_hi(unsigned r) {
  return __builtin_bit_cast(_Float16, (unsigned short)(r >> 16));
}

// ---- WMMA fragment loaders (CDNA5 16x16x32 f16 layouts) ----
// A matrix 16x32: lane L holds row m = L&15; VGPR v holds K pair at
//   k = 2*(v&3) + 8*(L>>4) + 16*(v>>2). rowptr = &src[m * stride].
__device__ __forceinline__ v16h a_frag_h(const _Float16* rowptr, int k0, int hi) {
  v16h a;
#pragma unroll
  for (int v = 0; v < 8; ++v) {
    int k = k0 + 2 * (v & 3) + 8 * hi + 16 * (v >> 2);
    unsigned r = *reinterpret_cast<const unsigned*>(rowptr + k);
    a[2 * v] = h_lo(r);
    a[2 * v + 1] = h_hi(r);
  }
  return a;
}
__device__ __forceinline__ v16h a_frag_f(const float* rowptr, int k0, int hi) {
  v16h a;
#pragma unroll
  for (int v = 0; v < 8; ++v) {
    int k = k0 + 2 * (v & 3) + 8 * hi + 16 * (v >> 2);
    float2 f = *reinterpret_cast<const float2*>(rowptr + k);
    a[2 * v] = (_Float16)f.x;
    a[2 * v + 1] = (_Float16)f.y;
  }
  return a;
}
// B matrix 32x16 (column-major fragment): lane L holds column n = L&15;
//   VGPR v holds K pair at k = 2*(v&3) + 8*(L>>4) + 16*(v>>2). rowptr = &src[n * stride]
//   where memory holds B^T rows (contiguous along K).
__device__ __forceinline__ v16h b_frag_h(const _Float16* rowptr, int k0, int hi) {
  v16h b;
#pragma unroll
  for (int v = 0; v < 8; ++v) {
    int k = k0 + 2 * (v & 3) + 8 * hi + 16 * (v >> 2);
    unsigned r = *reinterpret_cast<const unsigned*>(rowptr + k);
    b[2 * v] = h_lo(r);
    b[2 * v + 1] = h_hi(r);
  }
  return b;
}
__device__ __forceinline__ v16h b_frag_f(const float* rowptr, int k0, int hi) {
  v16h b;
#pragma unroll
  for (int v = 0; v < 8; ++v) {
    int k = k0 + 2 * (v & 3) + 8 * hi + 16 * (v >> 2);
    float2 f = *reinterpret_cast<const float2*>(rowptr + k);
    b[2 * v] = (_Float16)f.x;
    b[2 * v + 1] = (_Float16)f.y;
  }
  return b;
}

// ---- Projection GEMM: out[m,n] = f16( X[m,:] . W[n,:] + bias[n] )
// X: (16384 x 768) f32, W: (768 x 768) f32 row-major (rows = output features).
// TRANS==1: store as Vt[b][d][s] (for PV GEMM contiguity along keys).
template <int TRANS>
__global__ __launch_bounds__(128) void k_proj(
    const float* __restrict__ X, const float* __restrict__ W,
    const float* __restrict__ bias, _Float16* __restrict__ out) {
  const int lane = threadIdx.x;
  const int hi = lane >> 4, lo = lane & 15;
  const int n0 = (blockIdx.x * 4 + threadIdx.y) * 16;
  const int m0 = blockIdx.y * 16;

  const float* arow = X + (size_t)(m0 + lo) * DIM;
  const float* brow = W + (size_t)(n0 + lo) * DIM;

  v8f acc = {};
#pragma unroll 1
  for (int k0 = 0; k0 < DIM; k0 += 32) {
    __builtin_prefetch(arow + k0 + 128, 0, 0);
    v16h a = a_frag_f(arow, k0, hi);
    v16h b = b_frag_f(brow, k0, hi);
    acc = __builtin_amdgcn_wmma_f32_16x16x32_f16(false, a, false, b, (short)0,
                                                 acc, false, false);
  }
  const int n = n0 + lo;
  const float bv = bias[n];
  if (TRANS) {
    const int bb = m0 >> 10;           // tile never crosses a batch boundary
    _Float16* obase = out + (size_t)bb * (DIM * SEQ) + (size_t)n * SEQ;
    const int s0 = m0 & 1023;
#pragma unroll
    for (int v = 0; v < 8; ++v)
      obase[s0 + v + 8 * hi] = (_Float16)(acc[v] + bv);
  } else {
#pragma unroll
    for (int v = 0; v < 8; ++v) {
      int m = m0 + v + 8 * hi;
      out[(size_t)m * DIM + n] = (_Float16)(acc[v] + bv);
    }
  }
}

// ---- Final GEMM: out_f32[m,n] = H[m,:] . Wfc[n,:] + bfc[n], H f16.
__global__ __launch_bounds__(128) void k_final(
    const _Float16* __restrict__ H, const float* __restrict__ W,
    const float* __restrict__ bias, float* __restrict__ out) {
  const int lane = threadIdx.x;
  const int hi = lane >> 4, lo = lane & 15;
  const int n0 = (blockIdx.x * 4 + threadIdx.y) * 16;
  const int m0 = blockIdx.y * 16;

  const _Float16* arow = H + (size_t)(m0 + lo) * DIM;
  const float* brow = W + (size_t)(n0 + lo) * DIM;

  v8f acc = {};
#pragma unroll 1
  for (int k0 = 0; k0 < DIM; k0 += 32) {
    v16h a = a_frag_h(arow, k0, hi);
    v16h b = b_frag_f(brow, k0, hi);
    acc = __builtin_amdgcn_wmma_f32_16x16x32_f16(false, a, false, b, (short)0,
                                                 acc, false, false);
  }
  const int n = n0 + lo;
  const float bv = bias[n];
#pragma unroll
  for (int v = 0; v < 8; ++v) {
    int m = m0 + v + 8 * hi;
    out[(size_t)m * DIM + n] = acc[v] + bv;
  }
}

// ---- Fused dual cross-attention + residual + LayerNorm.
// One block per (batch, 16-query tile): 256 threads = 8 waves.
// LDS: sQ (16x768 f16, 24KB) + sP (16x1024 f16, 32KB); sO (16x768 f32, 48KB)
// aliases them after attention is done. Total static LDS < 64KB.
#define SMEM_BYTES 57344
__global__ __launch_bounds__(256) void k_attn(
    const _Float16* __restrict__ Q, const _Float16* __restrict__ Ka,
    const _Float16* __restrict__ Kb, const _Float16* __restrict__ Va,
    const _Float16* __restrict__ Vb, const float* __restrict__ xC,
    const float* __restrict__ gamma, const float* __restrict__ beta,
    _Float16* __restrict__ Hout) {
  __shared__ __align__(16) char smem[SMEM_BYTES];
  __shared__ float redA[16][16];
  __shared__ float redB[16][16];
  __shared__ float rowM[16];
  __shared__ float rowI[16];

  _Float16* sQ = reinterpret_cast<_Float16*>(smem);           // 16 x 768
  _Float16* sP = reinterpret_cast<_Float16*>(smem + 24576);   // 16 x 1024
  float* sO = reinterpret_cast<float*>(smem);                 // 16 x 768 (aliased)

  const int tid = threadIdx.x;
  const int lane = tid & 31, wave = tid >> 5;
  const int hi = lane >> 4, lo = lane & 15;
  const int qt = blockIdx.x;
  const int bb = qt >> 6;       // batch (64 query tiles per batch)
  const int tok0 = qt << 4;     // first global query token of tile
  const int ktok0 = bb << 10;   // first global key token of batch

  // Stage Q tile to LDS via CDNA5 async global->LDS DMA (no VGPR bounce):
  // 24 KB = 1536 x 16B chunks, 6 chunks per thread. Generic-pointer low 32
  // bits are the LDS byte offset (ISA aperture truncation rule).
  {
    const char* gsrc = reinterpret_cast<const char*>(Q + (size_t)tok0 * DIM);
    unsigned ldsbase = (unsigned)(unsigned long long)(void*)sQ;
#pragma unroll
    for (int i = 0; i < 6; ++i) {
      int idx = tid + 256 * i;  // 16-byte chunk index
      unsigned lds_addr = ldsbase + idx * 16;
      const char* gaddr = gsrc + (size_t)idx * 16;
      asm volatile("global_load_async_to_lds_b128 %0, %1, off"
                   :
                   : "v"(lds_addr), "v"(gaddr)
                   : "memory");
    }
    asm volatile("s_wait_asynccnt 0" ::: "memory");
  }
  __syncthreads();

  v8f zero = {};
  v8f accO[6];
#pragma unroll
  for (int t = 0; t < 6; ++t) accO[t] = zero;

  const int row = tid >> 4;  // 0..15 (softmax / layernorm phases)
  const int col = tid & 15;  // 16 threads per row

  for (int br = 0; br < 2; ++br) {
    const _Float16* K = br ? Kb : Ka;
    const _Float16* Vt = br ? Vb : Va;
    __syncthreads();  // previous reads of sP complete before overwrite

    // Scores: wave w computes key tiles [8w, 8w+8) of this batch.
    const _Float16* qrow = sQ + lo * DIM;
#pragma unroll 1
    for (int i = 0; i < 8; ++i) {
      int kt = wave * 8 + i;
      const _Float16* krow = K + (size_t)(ktok0 + kt * 16 + lo) * DIM;
      v8f c = zero;
#pragma unroll 1
      for (int k0 = 0; k0 < DIM; k0 += 32) {
        v16h a = a_frag_h(qrow, k0, hi);
        v16h b = b_frag_h(krow, k0, hi);
        c = __builtin_amdgcn_wmma_f32_16x16x32_f16(false, a, false, b, (short)0,
                                                   c, false, false);
      }
#pragma unroll
      for (int v = 0; v < 8; ++v) {
        int m = v + 8 * hi;
        sP[m * SEQ + kt * 16 + lo] = (_Float16)(c[v] * SM_SCALE);
      }
    }
    __syncthreads();

    // Row softmax over 1024 keys (16 threads/row, each owns 64 strided cols).
    {
      float mx = -3.0e38f;
      for (int j = col; j < SEQ; j += 16)
        mx = fmaxf(mx, (float)sP[row * SEQ + j]);
      redA[row][col] = mx;
      __syncthreads();
      if (col == 0) {
        float m2 = redA[row][0];
#pragma unroll
        for (int t = 1; t < 16; ++t) m2 = fmaxf(m2, redA[row][t]);
        rowM[row] = m2;
      }
      __syncthreads();
      float rmax = rowM[row];
      float s = 0.f;
      for (int j = col; j < SEQ; j += 16) {
        float e = __expf((float)sP[row * SEQ + j] - rmax);
        sP[row * SEQ + j] = (_Float16)e;
        s += e;
      }
      redA[row][col] = s;
      __syncthreads();
      if (col == 0) {
        float s2 = 0.f;
#pragma unroll
        for (int t = 0; t < 16; ++t) s2 += redA[row][t];
        rowI[row] = 1.f / s2;
      }
      __syncthreads();
      float inv = rowI[row];
      for (int j = col; j < SEQ; j += 16)
        sP[row * SEQ + j] = (_Float16)((float)sP[row * SEQ + j] * inv);
      __syncthreads();
    }

    // O += P @ V : wave w owns output columns [96w, 96w+96) (6 tiles of 16).
    const _Float16* prow = sP + lo * SEQ;
    const _Float16* vbase = Vt + (size_t)bb * (DIM * SEQ);
#pragma unroll 1
    for (int k0 = 0; k0 < SEQ; k0 += 32) {
      v16h a = a_frag_h(prow, k0, hi);
#pragma unroll
      for (int t = 0; t < 6; ++t) {
        int d = wave * 96 + t * 16 + lo;
        const _Float16* vrow = vbase + (size_t)d * SEQ;
        v16h b = b_frag_h(vrow, k0, hi);
        accO[t] = __builtin_amdgcn_wmma_f32_16x16x32_f16(
            false, a, false, b, (short)0, accO[t], false, false);
      }
    }
  }
  __syncthreads();  // all sQ/sP reads done; safe to alias with sO

  // Dump O accumulators to LDS in row-major f32.
#pragma unroll
  for (int t = 0; t < 6; ++t) {
    int d = wave * 96 + t * 16 + lo;
#pragma unroll
    for (int v = 0; v < 8; ++v) {
      int m = v + 8 * hi;
      sO[m * DIM + d] = accO[t][v];
    }
  }
  __syncthreads();

  // Residual + LayerNorm, store h as f16.
  const int token = tok0 + row;
  float sum = 0.f, sq = 0.f;
  for (int j = col; j < DIM; j += 16) {
    float v = sO[row * DIM + j] + xC[(size_t)token * DIM + j];
    sO[row * DIM + j] = v;
    sum += v;
    sq += v * v;
  }
  redA[row][col] = sum;
  redB[row][col] = sq;
  __syncthreads();
  if (col == 0) {
    float s = 0.f, q = 0.f;
#pragma unroll
    for (int t = 0; t < 16; ++t) { s += redA[row][t]; q += redB[row][t]; }
    float mu = s / (float)DIM;
    float var = q / (float)DIM - mu * mu;
    rowM[row] = mu;
    rowI[row] = rsqrtf(var + 1e-5f);
  }
  __syncthreads();
  {
    float mu = rowM[row], inv = rowI[row];
    for (int j = col; j < DIM; j += 16) {
      float v = (sO[row * DIM + j] - mu) * inv * gamma[j] + beta[j];
      Hout[(size_t)token * DIM + j] = (_Float16)v;
    }
  }
}

extern "C" void kernel_launch(void* const* d_in, const int* in_sizes, int n_in,
                              void* d_out, int out_size, void* d_ws,
                              size_t ws_size, hipStream_t stream) {
  const float* xA = (const float*)d_in[0];
  const float* xB = (const float*)d_in[1];
  const float* xC = (const float*)d_in[2];
  const float* Wq = (const float*)d_in[3];
  const float* bq = (const float*)d_in[4];
  const float* Wk = (const float*)d_in[5];
  const float* bk = (const float*)d_in[6];
  const float* Wv = (const float*)d_in[7];
  const float* bv = (const float*)d_in[8];
  const float* gamma = (const float*)d_in[9];
  const float* beta = (const float*)d_in[10];
  const float* Wfc = (const float*)d_in[11];
  const float* bfc = (const float*)d_in[12];

  // Workspace layout: 6 f16 chunks of 16384*768 elements (24 MB each, 144 MB).
  _Float16* ws = (_Float16*)d_ws;
  const size_t CH = (size_t)NTOK * DIM;
  _Float16* Q = ws + 0 * CH;
  _Float16* Ka = ws + 1 * CH;
  _Float16* Kb = ws + 2 * CH;
  _Float16* Va = ws + 3 * CH;  // stored transposed [b][d][s]
  _Float16* Vb = ws + 4 * CH;  // stored transposed [b][d][s]
  _Float16* H = ws + 5 * CH;

  dim3 gG(12, NTOK / 16);  // 12 n-groups (x4 tiles) x 1024 m-tiles
  dim3 gT(32, 4);          // 4 waves, one 16x16 tile each

  k_proj<0><<<gG, gT, 0, stream>>>(xC, Wq, bq, Q);
  k_proj<0><<<gG, gT, 0, stream>>>(xA, Wk, bk, Ka);
  k_proj<0><<<gG, gT, 0, stream>>>(xB, Wk, bk, Kb);
  k_proj<1><<<gG, gT, 0, stream>>>(xA, Wv, bv, Va);
  k_proj<1><<<gG, gT, 0, stream>>>(xB, Wv, bv, Vb);

  k_attn<<<NTOK / 16, 256, 0, stream>>>(Q, Ka, Kb, Va, Vb, xC, gamma, beta, H);

  k_final<<<gG, gT, 0, stream>>>(H, Wfc, bfc, (float*)d_out);
}
